// BottleneckBlock_79663053406454
// MI455X (gfx1250) — matrix operand order repaired and checked by reading, not compiled, not run on hardware
//
#include <hip/hip_runtime.h>
#include <hip/hip_fp16.h>
#include <math.h>

typedef __attribute__((ext_vector_type(16))) _Float16 v16h;
typedef __attribute__((ext_vector_type(8)))  float    v8f;

#define EMB   64
#define KBINS 2048
#define TLEN  8192
#define NBATCH 16
#define NT    (NBATCH*TLEN)       /* 131072 rows            */
#define NUMEL (NT*EMB)            /* 8388608 elements       */
#define MUF   0.99f

/* ---- workspace layout (bytes) ---- */
#define WS_KHALF 0                /* 2048*64 halfs  = 262144 */
#define WS_KNORM 262144           /* 2048 f32       =   8192 */
#define WS_KSUM  270336           /* 2048*64 f32    = 524288 */
#define WS_KELEM 794624           /* 2048 f32       =   8192 */
#define WS_SCAL  802816           /* 16 f32                  */
#define WS_ZERO_BYTES ((WS_SCAL + 64) - WS_KSUM)

/* ---- output layout (float elements, concatenated in return order) ---- */
#define O_XL       0
#define O_XD       131072
#define O_SC       (131072 + 8388608)   /* 8 scalars */
#define O_NEWK     (O_SC + 8)
#define O_NEWKSUM  (O_NEWK + 131072)
#define O_NEWKELEM (O_NEWKSUM + 131072)

/* ---------------- prep: f16 codebook copy + ||k||^2 ---------------- */
__global__ void vq_prep(const float* __restrict__ k,
                        _Float16* __restrict__ kh,
                        float* __restrict__ kn)
{
    int gi = blockIdx.x * blockDim.x + threadIdx.x;
    if (gi < KBINS * EMB) kh[gi] = (_Float16)k[gi];
    if (gi < KBINS) {
        float s = 0.f;
        #pragma unroll 8
        for (int c = 0; c < EMB; ++c) { float v = k[gi*EMB + c]; s += v*v; }
        kn[gi] = s;
    }
}

/* ---------------- main: WMMA distances + argmin + epilogue ---------------- */
__global__ __launch_bounds__(256, 1)
void vq_main(const float* __restrict__ x,
             const float* __restrict__ k,
             const _Float16* __restrict__ kh,
             const float* __restrict__ kn,
             float* __restrict__ out,
             float* __restrict__ ksum_acc,
             float* __restrict__ kelem_acc,
             float* __restrict__ scal)
{
    __shared__ _Float16 xh[256*72];     /* 36864 B : x tile, f16, padded rows  */
    __shared__ _Float16 kc[128*72];     /* 18432 B : codebook chunk, padded    */
    __shared__ float rn[256];
    __shared__ int   ridx[256];
    __shared__ float rdmin[256];

    const int tid  = threadIdx.x;
    const int lane = tid & 31;
    const int w    = tid >> 5;
    const int blk  = blockIdx.x;
    const int n    = blk >> 5;                 /* 32 blocks per batch entry */
    const int t0   = (blk & 31) * 256;
    const int grow0 = blk * 256;

    /* ---- load x tile: thread `tid` owns token row (t0+tid) entirely ---- */
    float sx = 0.f, sx2 = 0.f, rnorm = 0.f;
    {
        const float* xrow = x + ((long)n * EMB) * TLEN + t0 + tid;
        #pragma unroll 8
        for (int c = 0; c < EMB; ++c) {
            float v = xrow[(long)c * TLEN];      /* coalesced per c */
            sx += v; sx2 += v*v; rnorm += v*v;
            xh[tid*72 + c] = (_Float16)v;        /* transposed into LDS */
        }
        rn[tid] = rnorm;
    }
    __syncthreads();

    /* ---- build A fragments (2 row-tiles per wave, K split 0..31 / 32..63) */
    const int hi  = lane >> 4;
    const int nlo = lane & 15;
    const int rbase = w * 32;
    v16h a00, a01, a10, a11;
    {
        int lr0 = rbase + nlo;
        int lr1 = rbase + 16 + nlo;
        #pragma unroll
        for (int e = 0; e < 16; ++e) {
            int kk = hi * 16 + e;   /* contiguous K layout, matches B */
            a00[e] = xh[lr0*72 + kk];
            a01[e] = xh[lr0*72 + 32 + kk];
            a10[e] = xh[lr1*72 + kk];
            a11[e] = xh[lr1*72 + 32 + kk];
        }
    }
    float rn0[8], rn1[8];
    #pragma unroll
    for (int r = 0; r < 8; ++r) {
        rn0[r] = rn[rbase + hi*8 + r];
        rn1[r] = rn[rbase + 16 + hi*8 + r];
    }

    /* best holds m = ||k||^2 - 2<x,k>  (row-norm constant dropped: argmin-invariant) */
    float best0[8], best1[8]; int bi0[8], bi1[8];
    #pragma unroll
    for (int r = 0; r < 8; ++r) { best0[r] = 3.4e38f; best1[r] = 3.4e38f; bi0[r] = 0; bi1[r] = 0; }

    const unsigned kc_lds = (unsigned)(size_t)(&kc[0]);   /* LDS aperture: low 32b = byte offset */

    /* ---- sweep codebook in 16 chunks of 128 codes ---- */
    for (int chunk = 0; chunk < 16; ++chunk) {
        __syncthreads();                       /* previous chunk fully consumed */
        float knc[8];
        {   /* async-stage chunk directly into LDS (rows padded to 72 halfs).
               64-half code row = 8 aligned 16B granules -> no pad straddle.   */
            const _Float16* src = kh + chunk * 128 * EMB;
            #pragma unroll
            for (int j = 0; j < 4; ++j) {
                int g    = tid + j * 256;          /* 1024 granules of 16B */
                int code = g >> 3;
                int off  = (g & 7) * 8;
                unsigned ldsoff = kc_lds + (unsigned)((code*72 + off) * 2);
                const _Float16* gp = src + code*EMB + off;
                asm volatile("global_load_async_to_lds_b128 %0, %1, off"
                             :: "v"(ldsoff), "v"(gp) : "memory");
            }
            /* hoist this chunk's code norms into registers (hides L2 latency) */
            #pragma unroll
            for (int j = 0; j < 8; ++j) knc[j] = kn[chunk*128 + j*16 + nlo];
            if (chunk + 1 < 16)    /* warm L2/L0 for next chunk */
                __builtin_prefetch(kh + (chunk+1)*128*EMB + tid*32, 0, 1);
            asm volatile("s_wait_asynccnt 0" ::: "memory");
        }
        __syncthreads();                       /* chunk staged by all waves */

        for (int ct = 0; ct < 8; ++ct) {
            int code_l = ct*16 + nlo;
            int gcode  = chunk*128 + code_l;
            float knv  = knc[ct];
            v16h b0 = *(const v16h*)&kc[code_l*72 + hi*16];       /* K 0..31  */
            v16h b1 = *(const v16h*)&kc[code_l*72 + 32 + hi*16];  /* K 32..63 */
            v8f acc0 = {}; v8f acc1 = {};
            acc0 = __builtin_amdgcn_wmma_f32_16x16x32_f16(false, a00, false, b0, (short)0, acc0, false, false);
            acc0 = __builtin_amdgcn_wmma_f32_16x16x32_f16(false, a01, false, b1, (short)0, acc0, false, false);
            acc1 = __builtin_amdgcn_wmma_f32_16x16x32_f16(false, a10, false, b0, (short)0, acc1, false, false);
            acc1 = __builtin_amdgcn_wmma_f32_16x16x32_f16(false, a11, false, b1, (short)0, acc1, false, false);
            #pragma unroll
            for (int r = 0; r < 8; ++r) {
                float m0 = fmaf(-2.0f, acc0[r], knv);
                if (m0 < best0[r]) { best0[r] = m0; bi0[r] = gcode; }
                float m1 = fmaf(-2.0f, acc1[r], knv);
                if (m1 < best1[r]) { best1[r] = m1; bi1[r] = gcode; }
            }
        }
    }

    /* ---- argmin reduction across the 16 lanes holding one column group ---- */
    #pragma unroll
    for (int off = 1; off < 16; off <<= 1) {
        #pragma unroll
        for (int r = 0; r < 8; ++r) {
            float od = __shfl_xor(best0[r], off, 16);
            int   oi = __shfl_xor(bi0[r],  off, 16);
            if (od < best0[r] || (od == best0[r] && oi < bi0[r])) { best0[r] = od; bi0[r] = oi; }
            od = __shfl_xor(best1[r], off, 16);
            oi = __shfl_xor(bi1[r],  off, 16);
            if (od < best1[r] || (od == best1[r] && oi < bi1[r])) { best1[r] = od; bi1[r] = oi; }
        }
    }
    if (nlo == 0) {
        #pragma unroll
        for (int r = 0; r < 8; ++r) {
            ridx[rbase + hi*8 + r]       = bi0[r];
            rdmin[rbase + hi*8 + r]      = best0[r] + rn0[r];   /* restore row norm */
            ridx[rbase + 16 + hi*8 + r]  = bi1[r];
            rdmin[rbase + 16 + hi*8 + r] = best1[r] + rn1[r];
        }
    }
    __syncthreads();

    /* ---- epilogue: one thread per token ---- */
    float s_commit = 0.f;
    int   idx   = ridx[tid];
    float s_fit = rdmin[tid];
    float s_rowN = sqrtf(rn[tid]);
    {
        int gr = grow0 + tid;
        out[O_XL + gr] = (float)idx;
        const float* krow = k + idx * EMB;
        float* xd = out + O_XD + ((long)n * EMB) * TLEN + (t0 + tid);
        float* ks = ksum_acc + idx * EMB;
        #pragma unroll 8
        for (int c = 0; c < EMB; ++c) {
            float kv = krow[c];
            float xv = (float)xh[tid*72 + c];
            float dl = kv - xv;
            s_commit += dl*dl;
            xd[(long)c * TLEN] = kv;            /* coalesced per c */
            atomicAdd(&ks[c], xv);              /* segment sum     */
        }
        atomicAdd(&kelem_acc[idx], 1.0f);
    }

    /* ---- scalar partials: wave reduce, one atomic per wave ---- */
    auto wred = [&](float v) {
        for (int o = 16; o > 0; o >>= 1) v += __shfl_down(v, o, 32);
        return v;
    };
    float v;
    v = wred(sx);       if (lane == 0) atomicAdd(&scal[0], v);
    v = wred(sx2);      if (lane == 0) atomicAdd(&scal[1], v);
    v = wred(s_rowN);   if (lane == 0) atomicAdd(&scal[2], v);
    v = wred(s_fit);    if (lane == 0) atomicAdd(&scal[3], v);
    v = wred(s_commit); if (lane == 0) atomicAdd(&scal[4], v);
}

/* ---------------- per-bin EMA update ---------------- */
__global__ void vq_bins(const float* __restrict__ k,
                        const float* __restrict__ k_sum,
                        const float* __restrict__ k_elem,
                        const float* __restrict__ x,
                        const float* __restrict__ ksum_acc,
                        const float* __restrict__ kelem_acc,
                        float* __restrict__ out,
                        float* __restrict__ scal)
{
    int b = blockIdx.x * blockDim.x + threadIdx.x;
    if (b >= KBINS) return;
    float ke_raw = kelem_acc[b];
    float nke = MUF * k_elem[b] + (1.f - MUF) * ke_raw;
    out[O_NEWKELEM + b] = nke;
    float usage = (nke >= 1.0f) ? 1.f : 0.f;
    /* deterministic pseudo-permutation for random-restart rows */
    unsigned p = ((unsigned)b * 2654435761u) % (unsigned)NT;
    int pn = (int)(p >> 13), pt = (int)(p & 8191);
    const float* xr = x + ((long)pn * EMB) * TLEN + pt;
    float dk2 = 0.f;
    for (int c = 0; c < EMB; ++c) {
        float nks = MUF * k_sum[b*EMB + c] + (1.f - MUF) * ksum_acc[b*EMB + c];
        out[O_NEWKSUM + b*EMB + c] = nks;
        float krand = xr[(long)c * TLEN];
        float nk = (usage > 0.f) ? (nks / nke) : krand;
        out[O_NEWK + b*EMB + c] = nk;
        float d = nk - k[b*EMB + c];
        dk2 += d*d;
    }
    float pb  = ke_raw / (float)NT;
    atomicAdd(&scal[5], -pb * logf(pb + 1e-8f));
    atomicAdd(&scal[6], (ke_raw >= 1.0f) ? 1.f : 0.f);
    atomicAdd(&scal[7], usage);
    atomicAdd(&scal[8], dk2);
}

/* ---------------- final scalars ---------------- */
__global__ void vq_fin(const float* __restrict__ scal, float* __restrict__ out)
{
    if (threadIdx.x != 0 || blockIdx.x != 0) return;
    float commit = scal[4] / (float)NUMEL;
    float m  = scal[0] / (float)NUMEL;
    float ss = scal[1] - (float)NUMEL * m * m;
    float prenorm = sqrtf(fmaxf(ss, 0.f)) / sqrtf((float)NUMEL);
    out[O_SC + 0] = commit;
    out[O_SC + 1] = prenorm;
    out[O_SC + 2] = scal[3] / (float)NT;        /* fit         */
    out[O_SC + 3] = scal[2] / (float)NT;        /* prenorm_vec */
    out[O_SC + 4] = scal[5];                    /* entropy     */
    out[O_SC + 5] = scal[6];                    /* used_curr   */
    out[O_SC + 6] = scal[7];                    /* usage_sum   */
    out[O_SC + 7] = sqrtf(scal[8]) / sqrtf((float)(KBINS*EMB)); /* dk */
}

extern "C" void kernel_launch(void* const* d_in, const int* in_sizes, int n_in,
                              void* d_out, int out_size, void* d_ws, size_t ws_size,
                              hipStream_t stream)
{
    const float* x      = (const float*)d_in[0];
    const float* k      = (const float*)d_in[1];
    const float* k_sum  = (const float*)d_in[2];
    const float* k_elem = (const float*)d_in[3];
    float* out = (float*)d_out;
    char*  ws  = (char*)d_ws;
    _Float16* kh       = (_Float16*)(ws + WS_KHALF);
    float* kn          = (float*)(ws + WS_KNORM);
    float* ksum_acc    = (float*)(ws + WS_KSUM);
    float* kelem_acc   = (float*)(ws + WS_KELEM);
    float* scal        = (float*)(ws + WS_SCAL);

    hipMemsetAsync(ws + WS_KSUM, 0, WS_ZERO_BYTES, stream);
    vq_prep<<<512, 256, 0, stream>>>(k, kh, kn);
    vq_main<<<512, 256, 0, stream>>>(x, k, kh, kn, out, ksum_acc, kelem_acc, scal);
    vq_bins<<<8, 256, 0, stream>>>(k, k_sum, k_elem, x, ksum_acc, kelem_acc, out, scal);
    vq_fin<<<1, 32, 0, stream>>>(scal, out);
}
